// PGAHead_72206990180525
// MI455X (gfx1250) — matrix operand order repaired and checked by reading, not compiled, not run on hardware
//
#include <hip/hip_runtime.h>
#include <hip/hip_bf16.h>

typedef __attribute__((ext_vector_type(16))) _Float16 v16h;
typedef __attribute__((ext_vector_type(8)))  _Float16 v8h;
typedef __attribute__((ext_vector_type(8)))  float    v8f;

#define BN    2048
#define DD    512
#define PP    768
#define LLAY  4
#define KTOP  8
#define NPART 1024
#define BIGF  1e9f

// ---------------------------------------------------------------------------
// f32 -> f16 conversion
__global__ __launch_bounds__(256)
void k_cvt16(const float* __restrict__ src, _Float16* __restrict__ dst, size_t n) {
    size_t i = (size_t)blockIdx.x * blockDim.x + threadIdx.x;
    if (i < n) dst[i] = (_Float16)src[i];
}

// zero helpers
__global__ void k_zero_f32(float* p, int n) {
    int i = blockIdx.x * blockDim.x + threadIdx.x;
    if (i < n) p[i] = 0.0f;
}
__global__ void k_zero_i32(int* p, int n) {
    int i = blockIdx.x * blockDim.x + threadIdx.x;
    if (i < n) p[i] = 0;
}

// ---------------------------------------------------------------------------
// Row L2-normalize: one block per row
__global__ __launch_bounds__(256)
void k_rownorm(const float* __restrict__ X, float* __restrict__ Y, int cols) {
    __shared__ float sh[256];
    __shared__ float inv;
    const int row = blockIdx.x;
    const int tid = threadIdx.x;
    const float* xr = X + (size_t)row * cols;
    float s = 0.0f;
    for (int c = tid; c < cols; c += 256) { float v = xr[c]; s += v * v; }
    sh[tid] = s; __syncthreads();
    for (int k = 128; k > 0; k >>= 1) {
        if (tid < k) sh[tid] += sh[tid + k];
        __syncthreads();
    }
    if (tid == 0) inv = 1.0f / fmaxf(sqrtf(sh[0]), 1e-12f);
    __syncthreads();
    float* yr = Y + (size_t)row * cols;
    for (int c = tid; c < cols; c += 256) yr[c] = xr[c] * inv;
}

// ---------------------------------------------------------------------------
// f16-in / f32-out WMMA GEMM.  C[M,N] = A[M,K] * B  (+ epilogue)
//   bTrans == 0 : B stored row-major [K][N]
//   bTrans == 1 : B stored row-major [N][K]  (C = A * B^T)
//   mode 0: plain   mode 1: clip to [-1+1e-8, 1-1e-8]   mode 2: C += aux
//
// 256 threads = 8 waves; block tile 64(M) x 128(N); wave tile 32x32 = 2x2
// WMMA 16x16x32 tiles.  LDS tiles padded to 40-half rows (80B: 16B-aligned)
// so every fragment is fetched with ds_load_b128:
//   A lane frag = As[r][kh*8 .. +7]  ++  As[r][16+kh*8 .. +7]
//   B lane frag = Bt[c][kh*8 .. +7]  ++  Bt[c][16+kh*8 .. +7]   (B stored transposed in LDS)
__global__ __launch_bounds__(256)
void k_gemm_f16(const _Float16* __restrict__ Am, const _Float16* __restrict__ Bm,
                float* __restrict__ C, const float* __restrict__ aux,
                int M, int N, int K, int bTrans, int mode)
{
    __shared__ __align__(16) _Float16 As[64][40];
    __shared__ __align__(16) _Float16 Bt[128][40];
    const int tid  = threadIdx.x;
    const int wave = tid >> 5;
    const int lane = tid & 31;
    const int mw   = wave >> 2;        // 0..1 : M macro-tile
    const int nw   = wave & 3;         // 0..3 : N macro-tile
    const int m0   = blockIdx.y * 64;
    const int n0   = blockIdx.x * 128;

    v8f acc00 = {}, acc01 = {}, acc10 = {}, acc11 = {};

    // per-thread staging coordinates
    const int ar = tid >> 2;           // 0..63
    const int ac = (tid & 3) * 8;      // 0,8,16,24

    const int hrow = lane & 15;
    const int kh   = lane >> 4;

    for (int k0 = 0; k0 < K; k0 += 32) {
        // ---- stage A tile 64x32 (one b128 per thread) ----
        const _Float16* asrc = Am + (size_t)(m0 + ar) * K + k0 + ac;
        *(uint4*)&As[ar][ac] = *(const uint4*)asrc;
        if (k0 + 32 < K) __builtin_prefetch(asrc + 32, 0, 1);

        // ---- stage B tile as Bt[n][k] (128x32) ----
        if (!bTrans) {
            // global [k][n]: coalesced b128 reads, transpose on LDS store
            int k = tid >> 3;          // 0..31
            int n = (tid & 7) * 16;    // 0..112
            const _Float16* src = Bm + (size_t)(k0 + k) * N + n0 + n;
            _Float16 tmp[16];
            *(uint4*)&tmp[0] = *(const uint4*)src;
            *(uint4*)&tmp[8] = *(const uint4*)(src + 8);
            if (k0 + 32 < K) __builtin_prefetch(src + (size_t)32 * N, 0, 1);
            #pragma unroll
            for (int e = 0; e < 16; ++e) Bt[n + e][k] = tmp[e];
        } else {
            // global [n][k]: direct b128 copies
            int n = tid >> 1;          // 0..127
            int c = (tid & 1) * 16;    // 0 or 16
            const _Float16* src = Bm + (size_t)(n0 + n) * K + k0 + c;
            *(uint4*)&Bt[n][c]     = *(const uint4*)src;
            *(uint4*)&Bt[n][c + 8] = *(const uint4*)(src + 8);
            if (k0 + 32 < K) __builtin_prefetch(src + 32, 0, 1);
        }
        __syncthreads();

        // ---- fragments: all ds_load_b128 ----
        const int r0 = mw * 32 + hrow;
        const int r1 = r0 + 16;
        v8h a0lo = *(const v8h*)&As[r0][kh * 8];
        v8h a0hi = *(const v8h*)&As[r0][16 + kh * 8];
        v8h a1lo = *(const v8h*)&As[r1][kh * 8];
        v8h a1hi = *(const v8h*)&As[r1][16 + kh * 8];
        v16h a0 = __builtin_shufflevector(a0lo, a0hi,
                    0,1,2,3,4,5,6,7,8,9,10,11,12,13,14,15);
        v16h a1 = __builtin_shufflevector(a1lo, a1hi,
                    0,1,2,3,4,5,6,7,8,9,10,11,12,13,14,15);

        const int c0 = nw * 32 + hrow;
        const int c1 = c0 + 16;
        v8h b0lo = *(const v8h*)&Bt[c0][kh * 8];
        v8h b0hi = *(const v8h*)&Bt[c0][16 + kh * 8];
        v8h b1lo = *(const v8h*)&Bt[c1][kh * 8];
        v8h b1hi = *(const v8h*)&Bt[c1][16 + kh * 8];
        v16h b0 = __builtin_shufflevector(b0lo, b0hi,
                    0,1,2,3,4,5,6,7,8,9,10,11,12,13,14,15);
        v16h b1 = __builtin_shufflevector(b1lo, b1hi,
                    0,1,2,3,4,5,6,7,8,9,10,11,12,13,14,15);

        acc00 = __builtin_amdgcn_wmma_f32_16x16x32_f16(false, a0, false, b0,
                                                       (short)0, acc00, false, false);
        acc01 = __builtin_amdgcn_wmma_f32_16x16x32_f16(false, a0, false, b1,
                                                       (short)0, acc01, false, false);
        acc10 = __builtin_amdgcn_wmma_f32_16x16x32_f16(false, a1, false, b0,
                                                       (short)0, acc10, false, false);
        acc11 = __builtin_amdgcn_wmma_f32_16x16x32_f16(false, a1, false, b1,
                                                       (short)0, acc11, false, false);
        __syncthreads();
    }

    // ---- epilogue: f32 16x16 C/D layout per tile ----
    const int lcol  = lane & 15;
    const int lrow8 = (lane >> 4) * 8;
    #pragma unroll
    for (int mi = 0; mi < 2; ++mi) {
        #pragma unroll
        for (int ni = 0; ni < 2; ++ni) {
            v8f accv = (mi == 0) ? (ni == 0 ? acc00 : acc01)
                                 : (ni == 0 ? acc10 : acc11);
            const int mb = m0 + mw * 32 + mi * 16 + lrow8;
            const int cN = n0 + nw * 32 + ni * 16 + lcol;
            #pragma unroll
            for (int v = 0; v < 8; ++v) {
                size_t idx = (size_t)(mb + v) * N + cN;
                float val = accv[v];
                if (mode == 1)      val = fminf(fmaxf(val, -1.0f + 1e-8f), 1.0f - 1e-8f);
                else if (mode == 2) val = val + aux[idx];
                C[idx] = val;
            }
        }
    }
}

// ---------------------------------------------------------------------------
// Top-8 same-label neighbors per row (emulates the reference's masked top_k).
__global__ __launch_bounds__(256)
void k_topk(const float* __restrict__ S, const int* __restrict__ labels,
            int* __restrict__ topIdx, int B)
{
    __shared__ float sv[256];
    __shared__ int   si[256];
    __shared__ int   chosen[KTOP];
    const int row = blockIdx.x;
    const int tid = threadIdx.x;
    const int li  = labels[row];
    const float* Srow = S + (size_t)row * B;

    for (int t = 0; t < KTOP; ++t) {
        float best = -3.0e38f; int bi = B;
        for (int j = tid; j < B; j += 256) {
            bool used = false;
            for (int u = 0; u < t; ++u) used = used || (chosen[u] == j);
            if (used) continue;
            float v = (labels[j] == li)
                        ? (Srow[j] - (j == row ? BIGF : 0.0f))
                        : -BIGF;
            if (v > best || (v == best && j < bi)) { best = v; bi = j; }
        }
        sv[tid] = best; si[tid] = bi;
        __syncthreads();
        for (int k = 128; k > 0; k >>= 1) {
            if (tid < k) {
                if (sv[tid + k] > sv[tid] ||
                    (sv[tid + k] == sv[tid] && si[tid + k] < si[tid])) {
                    sv[tid] = sv[tid + k]; si[tid] = si[tid + k];
                }
            }
            __syncthreads();
        }
        if (tid == 0) chosen[t] = si[0];
        __syncthreads();
    }
    if (tid < KTOP) topIdx[row * KTOP + tid] = chosen[tid];
}

// A[i][j] = (i==j) ? 1e-6 : 0
__global__ __launch_bounds__(256)
void k_fillA(float* __restrict__ A, int B) {
    size_t p = (size_t)blockIdx.x * blockDim.x + threadIdx.x;
    if (p >= (size_t)B * B) return;
    int i = (int)(p / B), j = (int)(p % B);
    A[p] = (i == j) ? 1e-6f : 0.0f;
}

// symmetric scatter of alpha*clip(S,0) at top-k positions
__global__ __launch_bounds__(256)
void k_scatterA(const float* __restrict__ S, const int* __restrict__ topIdx,
                float* __restrict__ A, float alpha, int B)
{
    int g = blockIdx.x * blockDim.x + threadIdx.x;
    if (g >= B * KTOP) return;
    int i = g / KTOP;
    int j = topIdx[g];
    float v = alpha * fmaxf(S[(size_t)i * B + j], 0.0f);
    if (i == j) v += 1e-6f;
    A[(size_t)i * B + j] = v;
    A[(size_t)j * B + i] = v;
}

// dinv[i] = rsqrt(max(rowsum, 1e-8))
__global__ __launch_bounds__(256)
void k_rowsum_dinv(const float* __restrict__ A, float* __restrict__ dinv, int B) {
    __shared__ float sh[256];
    const int row = blockIdx.x;
    const int tid = threadIdx.x;
    const float* ar = A + (size_t)row * B;
    float s = 0.0f;
    for (int j = tid; j < B; j += 256) s += ar[j];
    sh[tid] = s; __syncthreads();
    for (int k = 128; k > 0; k >>= 1) {
        if (tid < k) sh[tid] += sh[tid + k];
        __syncthreads();
    }
    if (tid == 0) dinv[row] = rsqrtf(fmaxf(sh[0], 1e-8f));
}

__global__ __launch_bounds__(256)
void k_symnorm(float* __restrict__ A, const float* __restrict__ dinv, int B) {
    size_t p = (size_t)blockIdx.x * blockDim.x + threadIdx.x;
    if (p >= (size_t)B * B) return;
    int i = (int)(p / B), j = (int)(p % B);
    A[p] *= dinv[i] * dinv[j];
}

// ---------------------------------------------------------------------------
// Batchnorm column stats (one block per column)
__global__ __launch_bounds__(256)
void k_bnstats(const float* __restrict__ T, float* __restrict__ mu,
               float* __restrict__ var, int rows, int cols)
{
    __shared__ float s1[256], s2[256];
    const int c = blockIdx.x;
    const int tid = threadIdx.x;
    float a = 0.0f, b = 0.0f;
    for (int r = tid; r < rows; r += 256) {
        float v = T[(size_t)r * cols + c];
        a += v; b += v * v;
    }
    s1[tid] = a; s2[tid] = b; __syncthreads();
    for (int k = 128; k > 0; k >>= 1) {
        if (tid < k) { s1[tid] += s1[tid + k]; s2[tid] += s2[tid + k]; }
        __syncthreads();
    }
    if (tid == 0) {
        float m = s1[0] / (float)rows;
        mu[c]  = m;
        var[c] = s2[0] / (float)rows - m * m;
    }
}

__global__ __launch_bounds__(256)
void k_bn_relu_f16(const float* __restrict__ T, const float* __restrict__ mu,
                   const float* __restrict__ var, const float* __restrict__ g,
                   const float* __restrict__ b, _Float16* __restrict__ out,
                   int n, int cols)
{
    int p = blockIdx.x * blockDim.x + threadIdx.x;
    if (p >= n) return;
    int c = p % cols;
    float v = (T[p] - mu[c]) * rsqrtf(var[c] + 1e-5f) * g[c] + b[c];
    out[p] = (_Float16)fmaxf(v, 0.0f);
}

// ---------------------------------------------------------------------------
// Deterministic reductions: fixed-order block partials + single combiner
__global__ __launch_bounds__(256)
void k_sqdiff_partial(const float* __restrict__ X, const float* __restrict__ Y,
                      float* __restrict__ part, size_t n)
{
    __shared__ float sh[256];
    const int tid = threadIdx.x;
    float s = 0.0f;
    for (size_t p = (size_t)blockIdx.x * 256 + tid; p < n; p += (size_t)gridDim.x * 256) {
        float d = X[p] - Y[p];
        s += d * d;
    }
    sh[tid] = s; __syncthreads();
    for (int k = 128; k > 0; k >>= 1) {
        if (tid < k) sh[tid] += sh[tid + k];
        __syncthreads();
    }
    if (tid == 0) part[blockIdx.x] = sh[0];
}

__global__ __launch_bounds__(256)
void k_idea_partial(const float* __restrict__ Kmat, const int* __restrict__ labels,
                    const float* __restrict__ dinvI, float* __restrict__ part, int B)
{
    __shared__ float sh[256];
    const int tid = threadIdx.x;
    const size_t n = (size_t)B * B;
    float s = 0.0f;
    for (size_t p = (size_t)blockIdx.x * 256 + tid; p < n; p += (size_t)gridDim.x * 256) {
        int i = (int)(p / B), j = (int)(p % B);
        float aij = (i == j) ? 1.0f : (labels[i] == labels[j] ? 0.99f : 0.01f);
        float kid = aij * dinvI[i] * dinvI[j];
        float d = Kmat[p] - kid;
        s += d * d;
    }
    sh[tid] = s; __syncthreads();
    for (int k = 128; k > 0; k >>= 1) {
        if (tid < k) sh[tid] += sh[tid + k];
        __syncthreads();
    }
    if (tid == 0) part[blockIdx.x] = sh[0];
}

__global__ __launch_bounds__(256)
void k_accum(const float* __restrict__ part, int np, float scale, float* __restrict__ acc) {
    __shared__ float sh[256];
    const int tid = threadIdx.x;
    float s = 0.0f;
    for (int i = tid; i < np; i += 256) s += part[i];
    sh[tid] = s; __syncthreads();
    for (int k = 128; k > 0; k >>= 1) {
        if (tid < k) sh[tid] += sh[tid + k];
        __syncthreads();
    }
    if (tid == 0) acc[0] += sh[0] * scale;
}

// label histogram + ideal-kernel degree
__global__ void k_hist(const int* __restrict__ labels, int* __restrict__ hist, int B) {
    int i = blockIdx.x * blockDim.x + threadIdx.x;
    if (i < B) atomicAdd(&hist[labels[i]], 1);
}
__global__ void k_dinv_idea(const int* __restrict__ labels, const int* __restrict__ hist,
                            float* __restrict__ dinvI, int B) {
    int i = blockIdx.x * blockDim.x + threadIdx.x;
    if (i >= B) return;
    int c = hist[labels[i]];
    float d = 1.0f + 0.99f * (float)(c - 1) + 0.01f * (float)(B - c);
    dinvI[i] = rsqrtf(fmaxf(d, 1e-8f));
}

__global__ void k_final(const float* __restrict__ accs, float* __restrict__ out) {
    if (threadIdx.x == 0) {
        float lk = accs[0], lz = accs[1], li = accs[2];
        out[0] = lk; out[1] = lz; out[2] = li;
        out[3] = 128.0f * lk + 64.0f * lz + li;
    }
}

// ---------------------------------------------------------------------------
extern "C" void kernel_launch(void* const* d_in, const int* in_sizes, int n_in,
                              void* d_out, int out_size, void* d_ws, size_t ws_size,
                              hipStream_t stream)
{
    (void)in_sizes; (void)n_in; (void)out_size; (void)ws_size;
    const float* feats  = (const float*)d_in[0];
    const int*   labels = (const int*)  d_in[1];
    const float* fc1    = (const float*)d_in[2];
    const float* fc2    = (const float*)d_in[3];
    const float* gamma  = (const float*)d_in[4];
    const float* beta   = (const float*)d_in[5];
    const float* projw  = (const float*)d_in[6];
    float* out = (float*)d_out;

    char* base = (char*)d_ws;
    size_t off = 0;
    auto alloc = [&](size_t bytes) -> void* {
        off = (off + 255) & ~(size_t)255;
        void* p = base + off; off += bytes; return p;
    };

    float*    Xn    = (float*)   alloc((size_t)BN * DD * 4);
    _Float16* Xn16  = (_Float16*)alloc((size_t)BN * DD * 2);
    _Float16* X16   = (_Float16*)alloc((size_t)BN * DD * 2);
    float*    S     = (float*)   alloc((size_t)BN * BN * 4);
    float*    Amat  = (float*)   alloc((size_t)BN * BN * 4);
    _Float16* A16   = (_Float16*)alloc((size_t)BN * BN * 2);
    float*    Kb0   = (float*)   alloc((size_t)BN * BN * 4);
    float*    Kb1   = (float*)   alloc((size_t)BN * BN * 4);
    float*    H1    = (float*)   alloc((size_t)BN * DD * 4);
    _Float16* H1h   = (_Float16*)alloc((size_t)BN * DD * 2);
    float*    Tm    = (float*)   alloc((size_t)BN * DD * 4);
    _Float16* H2h   = (_Float16*)alloc((size_t)BN * DD * 2);
    float*    H3    = (float*)   alloc((size_t)BN * DD * 4);
    _Float16* H3h   = (_Float16*)alloc((size_t)BN * DD * 2);
    float*    Z     = (float*)   alloc((size_t)BN * DD * 4);
    _Float16* Z16   = (_Float16*)alloc((size_t)BN * DD * 2);
    _Float16* Wh    = (_Float16*)alloc((size_t)DD * DD * 2);
    _Float16* Ph    = (_Float16*)alloc((size_t)PP * DD * 2);
    float*    Zp    = (float*)   alloc((size_t)BN * PP * 4);
    float*    Zn0   = (float*)   alloc((size_t)BN * PP * 4);
    float*    Zn1   = (float*)   alloc((size_t)BN * PP * 4);
    int*      topIx = (int*)     alloc((size_t)BN * KTOP * 4);
    float*    dinv  = (float*)   alloc((size_t)BN * 4);
    float*    mu    = (float*)   alloc((size_t)DD * 4);
    float*    var   = (float*)   alloc((size_t)DD * 4);
    int*      hist  = (int*)     alloc(256 * 4);
    float*    dinvI = (float*)   alloc((size_t)BN * 4);
    float*    part  = (float*)   alloc((size_t)NPART * 4);
    float*    accs  = (float*)   alloc(16);

    const dim3 blk(256);
    const unsigned gBB = (unsigned)(((size_t)BN * BN + 255) / 256);
    const unsigned gBD = (unsigned)(((size_t)BN * DD + 255) / 256);

    k_zero_f32<<<1, 32, 0, stream>>>(accs, 4);
    k_cvt16<<<dim3((unsigned)(((size_t)PP * DD + 255) / 256)), blk, 0, stream>>>(
        projw, Ph, (size_t)PP * DD);

    float* Kbuf[2] = {Kb0, Kb1};
    float* Znb[2]  = {Zn0, Zn1};

    for (int i = 0; i < LLAY; ++i) {
        const float* Xi = feats + (size_t)i * BN * DD;
        const float alpha = 1.0f + 0.2f * (float)i / 3.0f;   // linspace(1.0,1.2,4)

        // cosine similarity S = Xn Xn^T (clipped)
        k_rownorm<<<dim3(BN), blk, 0, stream>>>(Xi, Xn, DD);
        k_cvt16<<<dim3(gBD), blk, 0, stream>>>(Xn, Xn16, (size_t)BN * DD);
        k_gemm_f16<<<dim3(BN / 128, BN / 64), blk, 0, stream>>>(
            Xn16, Xn16, S, nullptr, BN, BN, DD, 1, 1);

        // intra-class top-8 adjacency (beta==0 -> inter branch dead)
        k_topk<<<dim3(BN), blk, 0, stream>>>(S, labels, topIx, BN);
        k_fillA<<<dim3(gBB), blk, 0, stream>>>(Amat, BN);
        k_scatterA<<<dim3((BN * KTOP + 255) / 256), blk, 0, stream>>>(
            S, topIx, Amat, alpha, BN);
        k_rowsum_dinv<<<dim3(BN), blk, 0, stream>>>(Amat, dinv, BN);
        k_symnorm<<<dim3(gBB), blk, 0, stream>>>(Amat, dinv, BN);
        k_cvt16<<<dim3(gBB), blk, 0, stream>>>(Amat, A16, (size_t)BN * BN);

        // diffusion kernel K = A_norm @ A_norm
        float* Kcur  = Kbuf[i & 1];
        float* Kprev = Kbuf[(i + 1) & 1];
        k_gemm_f16<<<dim3(BN / 128, BN / 64), blk, 0, stream>>>(
            A16, A16, Kcur, nullptr, BN, BN, BN, 0, 0);

        // gam: Z = A @ relu(BN(A @ (X W1^T))) W2^T + X
        k_cvt16<<<dim3(gBD), blk, 0, stream>>>(Xi, X16, (size_t)BN * DD);
        k_cvt16<<<dim3((unsigned)(((size_t)DD * DD + 255) / 256)), blk, 0, stream>>>(
            fc1 + (size_t)i * DD * DD, Wh, (size_t)DD * DD);
        k_gemm_f16<<<dim3(DD / 128, BN / 64), blk, 0, stream>>>(
            X16, Wh, H1, nullptr, BN, DD, DD, 1, 0);
        k_cvt16<<<dim3(gBD), blk, 0, stream>>>(H1, H1h, (size_t)BN * DD);
        k_gemm_f16<<<dim3(DD / 128, BN / 64), blk, 0, stream>>>(
            A16, H1h, Tm, nullptr, BN, DD, BN, 0, 0);
        k_bnstats<<<dim3(DD), blk, 0, stream>>>(Tm, mu, var, BN, DD);
        k_bn_relu_f16<<<dim3(gBD), blk, 0, stream>>>(
            Tm, mu, var, gamma + (size_t)i * DD, beta + (size_t)i * DD, H2h, BN * DD, DD);
        k_cvt16<<<dim3((unsigned)(((size_t)DD * DD + 255) / 256)), blk, 0, stream>>>(
            fc2 + (size_t)i * DD * DD, Wh, (size_t)DD * DD);
        k_gemm_f16<<<dim3(DD / 128, BN / 64), blk, 0, stream>>>(
            H2h, Wh, H3, nullptr, BN, DD, DD, 1, 0);
        k_cvt16<<<dim3(gBD), blk, 0, stream>>>(H3, H3h, (size_t)BN * DD);
        k_gemm_f16<<<dim3(DD / 128, BN / 64), blk, 0, stream>>>(
            A16, H3h, Z, Xi, BN, DD, BN, 0, 2);          // + residual X

        // projected, row-normalized embedding
        k_cvt16<<<dim3(gBD), blk, 0, stream>>>(Z, Z16, (size_t)BN * DD);
        k_gemm_f16<<<dim3(PP / 128, BN / 64), blk, 0, stream>>>(
            Z16, Ph, Zp, nullptr, BN, PP, DD, 1, 0);
        k_rownorm<<<dim3(BN), blk, 0, stream>>>(Zp, Znb[i & 1], PP);

        if (i >= 1) {
            k_sqdiff_partial<<<dim3(NPART), blk, 0, stream>>>(
                Kprev, Kcur, part, (size_t)BN * BN);
            k_accum<<<1, blk, 0, stream>>>(part, NPART,
                1.0f / ((float)BN * (float)BN), accs + 0);
            k_sqdiff_partial<<<dim3(NPART), blk, 0, stream>>>(
                Znb[(i + 1) & 1], Znb[i & 1], part, (size_t)BN * PP);
            k_accum<<<1, blk, 0, stream>>>(part, NPART,
                1.0f / ((float)BN * (float)PP), accs + 1);
        }
        if (i == LLAY - 1) {
            k_zero_i32<<<1, 256, 0, stream>>>(hist, 256);
            k_hist<<<dim3((BN + 255) / 256), blk, 0, stream>>>(labels, hist, BN);
            k_dinv_idea<<<dim3((BN + 255) / 256), blk, 0, stream>>>(labels, hist, dinvI, BN);
            k_idea_partial<<<dim3(NPART), blk, 0, stream>>>(Kcur, labels, dinvI, part, BN);
            k_accum<<<1, blk, 0, stream>>>(part, NPART,
                1.0f / ((float)BN * (float)BN), accs + 2);
        }
    }

    k_final<<<1, 32, 0, stream>>>(accs, out);
}
